// Model_12678743458478
// MI455X (gfx1250) — matrix operand order repaired and checked by reading, not compiled, not run on hardware
//
#include <hip/hip_runtime.h>
#include <hip/hip_bf16.h>

// ---------------- model dims ----------------
constexpr int Bc = 32, Lc = 512, Ec = 640, Nc = 200000, Kc = 3, NLc = 12, Hc = 8;
constexpr int MIDc = 1280, FFc = 2048, CLc = 6, MLc = 3;
constexpr float RATIOc = 0.2f;
constexpr float BN_SCALE = 0.9999950000374997f;     // 1/sqrt(1+1e-5)
constexpr float INV_SQRT_DK = 0.11180339887498949f; // 1/sqrt(80)
constexpr int TKB = 512;                            // top-k candidate blocks
constexpr int LDST = 40;                            // padded LDS tile row stride (bf16 elems)

// ---------------- WMMA helpers ----------------
typedef __attribute__((ext_vector_type(16))) __bf16 v16bf;
typedef __attribute__((ext_vector_type(8)))  __bf16 v8bf;
typedef __attribute__((ext_vector_type(8)))  float  v8f;

// A fragment: 16x32 (MxK) bf16, ISA layout (interleaved K groups of 8). 16B LDS loads.
__device__ __forceinline__ v16bf frag_load_a(const __bf16* p0, int stride, int lane) {
  int r = lane & 15, kg = (lane >> 4); // 0/1
  const v8bf* p = (const v8bf*)(p0 + r * stride);
  v8bf lo = p[kg];
  v8bf hi = p[2 + kg];
  v16bf f;
#pragma unroll
  for (int e = 0; e < 8; ++e) { f[e] = lo[e]; f[8 + e] = hi[e]; }
  return f;
}
// B fragment: 32x16 (KxN); lane holds column n=lane&15, ISA layout (interleaved K groups of 8).
__device__ __forceinline__ v16bf frag_load_b(const __bf16* p0, int stride, int lane) {
  int r = lane & 15, kg = (lane >> 4); // 0/1
  const v8bf* p = (const v8bf*)(p0 + r * stride);
  v8bf lo = p[kg];
  v8bf hi = p[2 + kg];
  v16bf f;
#pragma unroll
  for (int e = 0; e < 8; ++e) { f[e] = lo[e]; f[8 + e] = hi[e]; }
  return f;
}
#define WMMA_BF16(Af, Bf, Cf) \
  __builtin_amdgcn_wmma_f32_16x16x32_bf16(false, (Af), false, (Bf), (short)0, (Cf), false, false)

__device__ __forceinline__ float wave_sum(float v) {
#pragma unroll
  for (int m = 16; m >= 1; m >>= 1) v += __shfl_xor(v, m, 32);
  return v;
}
__device__ __forceinline__ float wave_max(float v) {
#pragma unroll
  for (int m = 16; m >= 1; m >>= 1) v = fmaxf(v, __shfl_xor(v, m, 32));
  return v;
}

// ---------------- conversion / packing ----------------
__global__ void k_f32_to_bf16(const float* __restrict__ in, __bf16* __restrict__ out, long n) {
  long i = (long)blockIdx.x * blockDim.x + threadIdx.x;
  if (i < n) out[i] = (__bf16)in[i];
}

// conv weight (O,I,T) f32 row-major -> bf16 packed [t][o][i]
__global__ void k_pack_w(const float* __restrict__ in, __bf16* __restrict__ out,
                         int O, int I, int T) {
  long i = (long)blockIdx.x * blockDim.x + threadIdx.x;
  long n = (long)O * I * T;
  if (i >= n) return;
  int t = (int)(i % T); long rem = i / T;
  int ii = (int)(rem % I); int o = (int)(rem / I);
  out[(size_t)t * O * I + (size_t)o * I + ii] = (__bf16)in[i];
}

// embeddings (B,L,E) f32 -> xT (B,E,L) bf16
__global__ void k_emb_to_xT(const float* __restrict__ emb, __bf16* __restrict__ xT) {
  long i = (long)blockIdx.x * blockDim.x + threadIdx.x;
  long total = (long)Bc * Ec * Lc;
  if (i >= total) return;
  int l = (int)(i % Lc); long r = i / Lc;
  int e = (int)(r % Ec); int b = (int)(r / Ec);
  xT[i] = (__bf16)emb[((size_t)b * Lc + l) * Ec + e];
}

// ---------------- fused WMMA conv stage ----------------
// Y = relu(bn(conv_T(X)+cb)) + (conv1(X)+rb); X:(B,I,L) bf16, Wp packed [t][o][i]
// Block tile: 64(o) x 128(l); 8 waves = 2(o) x 4(l); wave tile 32x32 (2x2 frags).
__global__ __launch_bounds__(256) void k_conv_wmma(
    const __bf16* __restrict__ X, const __bf16* __restrict__ Wp,
    const __bf16* __restrict__ Rp, const float* __restrict__ cbias,
    const float* __restrict__ rbias, const float* __restrict__ bng,
    const float* __restrict__ bnb, __bf16* __restrict__ Ybf,
    float* __restrict__ Yf, int I, int O, int L, int T, int pad) {
  __shared__ __bf16 Wt[64 * LDST];
  __shared__ __bf16 Rt[64 * LDST];
  __shared__ __bf16 Xt[128 * LDST]; // [pos][chan]
  int tid = threadIdx.x, lane = tid & 31, wave = tid >> 5;
  int wo = (wave >> 2) * 32; // o offset within tile
  int wl = (wave & 3) * 32;  // l offset within tile
  int l0 = blockIdx.x * 128, o0 = blockIdx.y * 64, b = blockIdx.z;
  v8f accM[2][2], accR[2][2];
#pragma unroll
  for (int mi = 0; mi < 2; ++mi)
#pragma unroll
    for (int ni = 0; ni < 2; ++ni) {
      accM[mi][ni] = (v8f){0.f, 0.f, 0.f, 0.f, 0.f, 0.f, 0.f, 0.f};
      accR[mi][ni] = (v8f){0.f, 0.f, 0.f, 0.f, 0.f, 0.f, 0.f, 0.f};
    }
  for (int t = 0; t < T; ++t) {
    for (int i0 = 0; i0 < I; i0 += 32) {
      { // W tile 64x32: 8 elems/thread (contiguous)
        int oo = tid >> 2, ii = (tid & 3) * 8;
        const __bf16* src = Wp + (size_t)t * O * I + (size_t)(o0 + oo) * I + i0 + ii;
        __bf16* dst = Wt + oo * LDST + ii;
#pragma unroll
        for (int e = 0; e < 8; ++e) dst[e] = src[e];
      }
      if (t == pad) { // residual tile
        int oo = tid >> 2, ii = (tid & 3) * 8;
        const __bf16* src = Rp + (size_t)(o0 + oo) * I + i0 + ii;
        __bf16* dst = Rt + oo * LDST + ii;
#pragma unroll
        for (int e = 0; e < 8; ++e) dst[e] = src[e];
      }
      { // X tile transposed: Xt[pos][chan]; 16 positions per thread of one channel
        int ii = tid >> 3;
        int nb = (tid & 7) * 16;
        const __bf16* src = X + ((size_t)b * I + (i0 + ii)) * L;
#pragma unroll
        for (int e = 0; e < 16; ++e) {
          int n = nb + e;
          int p = l0 + n + t - pad;
          __bf16 v = (p >= 0 && p < L) ? src[p] : (__bf16)0.0f;
          Xt[n * LDST + ii] = v;
        }
      }
      __syncthreads();
      v16bf a0 = frag_load_a(Wt + wo * LDST, LDST, lane);
      v16bf a1 = frag_load_a(Wt + (wo + 16) * LDST, LDST, lane);
      v16bf b0 = frag_load_b(Xt + wl * LDST, LDST, lane);
      v16bf b1 = frag_load_b(Xt + (wl + 16) * LDST, LDST, lane);
      accM[0][0] = WMMA_BF16(a0, b0, accM[0][0]);
      accM[0][1] = WMMA_BF16(a0, b1, accM[0][1]);
      accM[1][0] = WMMA_BF16(a1, b0, accM[1][0]);
      accM[1][1] = WMMA_BF16(a1, b1, accM[1][1]);
      if (t == pad) {
        v16bf r0 = frag_load_a(Rt + wo * LDST, LDST, lane);
        v16bf r1 = frag_load_a(Rt + (wo + 16) * LDST, LDST, lane);
        accR[0][0] = WMMA_BF16(r0, b0, accR[0][0]);
        accR[0][1] = WMMA_BF16(r0, b1, accR[0][1]);
        accR[1][0] = WMMA_BF16(r1, b0, accR[1][0]);
        accR[1][1] = WMMA_BF16(r1, b1, accR[1][1]);
      }
      __syncthreads();
    }
  }
#pragma unroll
  for (int mi = 0; mi < 2; ++mi)
#pragma unroll
    for (int ni = 0; ni < 2; ++ni) {
      int lpos = l0 + wl + ni * 16 + (lane & 15);
      int ob = o0 + wo + mi * 16 + ((lane >> 4) << 3);
#pragma unroll
      for (int v = 0; v < 8; ++v) {
        int o = ob + v;
        float m = accM[mi][ni][v] + cbias[o];
        m = m * bng[o] * BN_SCALE + bnb[o];
        m = fmaxf(m, 0.f);
        float val = m + accR[mi][ni][v] + rbias[o];
        size_t oidx = ((size_t)b * O + o) * L + lpos;
        if (Ybf) Ybf[oidx] = (__bf16)val;
        if (Yf) Yf[oidx] = val;
      }
    }
}

// ---------------- generic WMMA GEMM: C(MxN)=A(MxK)@W(NxK)^T + bias ----------------
// Block tile: 128(m) x 64(n); 8 waves = 4(m) x 2(n); wave tile 32x32.
__global__ __launch_bounds__(256) void k_gemm_wmma(
    const __bf16* __restrict__ A, const __bf16* __restrict__ Wb,
    const float* __restrict__ bias, float* __restrict__ C, int M, int N, int Kd) {
  __shared__ __bf16 At[128 * LDST];
  __shared__ __bf16 Wt[64 * LDST];
  int tid = threadIdx.x, lane = tid & 31, wave = tid >> 5;
  int wm = (wave >> 1) * 32, wn = (wave & 1) * 32;
  int n0 = blockIdx.x * 64, m0 = blockIdx.y * 128;
  v8f acc[2][2];
#pragma unroll
  for (int mi = 0; mi < 2; ++mi)
#pragma unroll
    for (int ni = 0; ni < 2; ++ni) acc[mi][ni] = (v8f){0.f, 0.f, 0.f, 0.f, 0.f, 0.f, 0.f, 0.f};
  for (int k0 = 0; k0 < Kd; k0 += 32) {
    { // A tile 128x32: 16/thread
      int mm = tid >> 1, kk = (tid & 1) * 16;
      const __bf16* src = A + (size_t)(m0 + mm) * Kd + k0 + kk;
      __bf16* dst = At + mm * LDST + kk;
#pragma unroll
      for (int e = 0; e < 16; ++e) dst[e] = src[e];
    }
    { // W tile 64x32: 8/thread
      int nn = tid >> 2, kk = (tid & 3) * 8;
      const __bf16* src = Wb + (size_t)(n0 + nn) * Kd + k0 + kk;
      __bf16* dst = Wt + nn * LDST + kk;
#pragma unroll
      for (int e = 0; e < 8; ++e) dst[e] = src[e];
    }
    __syncthreads();
    v16bf a0 = frag_load_a(At + wm * LDST, LDST, lane);
    v16bf a1 = frag_load_a(At + (wm + 16) * LDST, LDST, lane);
    v16bf b0 = frag_load_b(Wt + wn * LDST, LDST, lane); // B[k][n]=W[n][k]
    v16bf b1 = frag_load_b(Wt + (wn + 16) * LDST, LDST, lane);
    acc[0][0] = WMMA_BF16(a0, b0, acc[0][0]);
    acc[0][1] = WMMA_BF16(a0, b1, acc[0][1]);
    acc[1][0] = WMMA_BF16(a1, b0, acc[1][0]);
    acc[1][1] = WMMA_BF16(a1, b1, acc[1][1]);
    __syncthreads();
  }
#pragma unroll
  for (int mi = 0; mi < 2; ++mi)
#pragma unroll
    for (int ni = 0; ni < 2; ++ni) {
      int n = n0 + wn + ni * 16 + (lane & 15);
      int mb = m0 + wm + mi * 16 + ((lane >> 4) << 3);
#pragma unroll
      for (int v = 0; v < 8; ++v) {
        int m = mb + v;
        C[(size_t)m * N + n] = acc[mi][ni][v] + (bias ? bias[n] : 0.f);
      }
    }
}

// ---------------- small VALU GEMM: C=act(A@W^T+b), A row m at A+m*lda ----------------
__global__ void k_gemm_small(const float* __restrict__ A, const float* __restrict__ Wm,
                             const float* __restrict__ bias, float* __restrict__ C,
                             int M, int N, int Kd, int lda, int act) {
  int i = blockIdx.x * blockDim.x + threadIdx.x;
  if (i >= M * N) return;
  int n = i % N, m = i / N;
  const float* a = A + (size_t)m * lda;
  const float* w = Wm + (size_t)n * Kd;
  float s = bias ? bias[n] : 0.f;
  for (int k = 0; k < Kd; ++k) s += a[k] * w[k];
  if (act == 1) s = tanhf(s);
  else if (act == 2) s = fmaxf(s, 0.f);
  C[(size_t)m * N + n] = s;
}

// ---------------- pooling, LN ----------------
__global__ __launch_bounds__(256) void k_pool(const float* __restrict__ V,
                                              const float* __restrict__ W,
                                              float* __restrict__ xs, float* __restrict__ xm) {
  int wave = threadIdx.x >> 5, lane = threadIdx.x & 31;
  int row = blockIdx.x * 8 + wave;
  if (row >= Bc * Ec) return;
  const float* vp = V + (size_t)row * Lc;
  const float* wp = W + (size_t)row * Lc;
  float s = 0.f, mx = -3.4e38f;
  for (int l = lane; l < Lc; l += 32) { float v = vp[l]; s += v * wp[l]; mx = fmaxf(mx, v); }
  s = wave_sum(s); mx = wave_max(mx);
  if (lane == 0) { xs[row] = s; xm[row] = mx; }
}

__global__ __launch_bounds__(256) void k_ln_to_h(const float* __restrict__ xs,
                                                 const float* __restrict__ xm,
                                                 const float* __restrict__ g,
                                                 const float* __restrict__ bb,
                                                 float* __restrict__ h) {
  int b = blockIdx.x, tid = threadIdx.x;
  __shared__ float red[256];
  for (int part = 0; part < 2; ++part) {
    const float* src = (part == 0 ? xs : xm) + (size_t)b * Ec;
    float s = 0.f;
    for (int e = tid; e < Ec; e += 256) s += src[e];
    red[tid] = s; __syncthreads();
    for (int st = 128; st > 0; st >>= 1) { if (tid < st) red[tid] += red[tid + st]; __syncthreads(); }
    float mu = red[0] / Ec; __syncthreads();
    float v = 0.f;
    for (int e = tid; e < Ec; e += 256) { float d = src[e] - mu; v += d * d; }
    red[tid] = v; __syncthreads();
    for (int st = 128; st > 0; st >>= 1) { if (tid < st) red[tid] += red[tid + st]; __syncthreads(); }
    float inv = rsqrtf(red[0] / Ec + 1e-5f); __syncthreads();
    for (int e = tid; e < Ec; e += 256)
      h[(size_t)b * 2 * Ec + part * Ec + e] = (src[e] - mu) * inv * g[e] + bb[e];
  }
}

__global__ __launch_bounds__(256) void k_ln_res(float* __restrict__ x,
                                                const float* __restrict__ y,
                                                const float* __restrict__ g,
                                                const float* __restrict__ bb) {
  int b = blockIdx.x, tid = threadIdx.x;
  __shared__ float row[Ec];
  __shared__ float red[256];
  for (int e = tid; e < Ec; e += 256) row[e] = x[(size_t)b * Ec + e] + y[(size_t)b * Ec + e];
  __syncthreads();
  float s = 0.f;
  for (int e = tid; e < Ec; e += 256) s += row[e];
  red[tid] = s; __syncthreads();
  for (int st = 128; st > 0; st >>= 1) { if (tid < st) red[tid] += red[tid + st]; __syncthreads(); }
  float mu = red[0] / Ec; __syncthreads();
  float v = 0.f;
  for (int e = tid; e < Ec; e += 256) { float d = row[e] - mu; v += d * d; }
  red[tid] = v; __syncthreads();
  for (int st = 128; st > 0; st >>= 1) { if (tid < st) red[tid] += red[tid + st]; __syncthreads(); }
  float inv = rsqrtf(red[0] / Ec + 1e-5f); __syncthreads();
  for (int e = tid; e < Ec; e += 256) x[(size_t)b * Ec + e] = (row[e] - mu) * inv * g[e] + bb[e];
}

// ---------------- queries / norms ----------------
__global__ void k_qmean(const float* __restrict__ emb, float* __restrict__ q) {
  int b = blockIdx.x;
  for (int e = threadIdx.x; e < Ec; e += blockDim.x) {
    float s = 0.f;
    for (int l = 0; l < Lc; ++l) s += emb[((size_t)b * Lc + l) * Ec + e];
    q[(size_t)b * Ec + e] = s / (float)Lc;
  }
}

__global__ __launch_bounds__(256) void k_rownorm(const float* __restrict__ A,
                                                 float* __restrict__ nrm, int cols) {
  int r = blockIdx.x, tid = threadIdx.x;
  __shared__ float red[256];
  float s = 0.f;
  for (int c = tid; c < cols; c += 256) { float v = A[(size_t)r * cols + c]; s += v * v; }
  red[tid] = s; __syncthreads();
  for (int st = 128; st > 0; st >>= 1) { if (tid < st) red[tid] += red[tid + st]; __syncthreads(); }
  if (tid == 0) nrm[r] = sqrtf(red[0]);
}

// ---------------- sims + per-block top3 ----------------
__global__ __launch_bounds__(256) void k_sims_topk(const float* __restrict__ db,
                                                   const float* __restrict__ q,
                                                   const float* __restrict__ qn,
                                                   float* __restrict__ cs,
                                                   int* __restrict__ ci) {
  __shared__ float row[Ec];
  int tid = threadIdx.x, lane = tid & 31, wave = tid >> 5;
  int rows_per = (Nc + gridDim.x - 1) / gridDim.x;
  int r0 = blockIdx.x * rows_per;
  int r1 = r0 + rows_per; if (r1 > Nc) r1 = Nc;
  float bs[4][3]; int bi[4][3];
#pragma unroll
  for (int a = 0; a < 4; ++a)
#pragma unroll
    for (int k = 0; k < 3; ++k) { bs[a][k] = -3.4e38f; bi[a][k] = 0; }
  for (int r = r0; r < r1; ++r) {
    const float* dr = db + (size_t)r * Ec;
    if (r + 8 < r1) __builtin_prefetch(db + (size_t)(r + 8) * Ec + tid, 0, 0);
    for (int e = tid; e < Ec; e += 256) row[e] = dr[e];
    __syncthreads();
    float sq = 0.f;
    for (int e = lane; e < Ec; e += 32) sq += row[e] * row[e];
    sq = wave_sum(sq);
    float rinv = rsqrtf(sq);
#pragma unroll
    for (int a = 0; a < 4; ++a) {
      int qi = wave * 4 + a;
      const float* qr = q + (size_t)qi * Ec;
      float d = 0.f;
      for (int e = lane; e < Ec; e += 32) d += row[e] * qr[e];
      d = wave_sum(d);
      if (lane == 0) {
        float sim = d * rinv / qn[qi];
        if (sim > bs[a][0]) {
          bs[a][2] = bs[a][1]; bi[a][2] = bi[a][1];
          bs[a][1] = bs[a][0]; bi[a][1] = bi[a][0];
          bs[a][0] = sim; bi[a][0] = r;
        } else if (sim > bs[a][1]) {
          bs[a][2] = bs[a][1]; bi[a][2] = bi[a][1];
          bs[a][1] = sim; bi[a][1] = r;
        } else if (sim > bs[a][2]) { bs[a][2] = sim; bi[a][2] = r; }
      }
    }
    __syncthreads();
  }
  if (lane == 0) {
    for (int a = 0; a < 4; ++a) {
      int qi = wave * 4 + a;
      size_t base = ((size_t)blockIdx.x * Bc + qi) * 3;
      for (int k = 0; k < 3; ++k) { cs[base + k] = bs[a][k]; ci[base + k] = bi[a][k]; }
    }
  }
}

__global__ __launch_bounds__(256) void k_topk_merge(const float* __restrict__ cs,
                                                    const int* __restrict__ ci,
                                                    float* __restrict__ ts,
                                                    int* __restrict__ ti, int nblocks) {
  int qi = blockIdx.x, tid = threadIdx.x;
  __shared__ float ss[256 * 3];
  __shared__ int si[256 * 3];
  float bs[3] = {-3.4e38f, -3.4e38f, -3.4e38f}; int bi[3] = {0, 0, 0};
  auto ins = [&](float s, int id) {
    if (s > bs[0] || (s == bs[0] && id < bi[0])) {
      bs[2] = bs[1]; bi[2] = bi[1]; bs[1] = bs[0]; bi[1] = bi[0]; bs[0] = s; bi[0] = id;
    } else if (s > bs[1] || (s == bs[1] && id < bi[1])) {
      bs[2] = bs[1]; bi[2] = bi[1]; bs[1] = s; bi[1] = id;
    } else if (s > bs[2] || (s == bs[2] && id < bi[2])) { bs[2] = s; bi[2] = id; }
  };
  for (int blk = tid; blk < nblocks; blk += 256) {
    size_t base = ((size_t)blk * Bc + qi) * 3;
    for (int k = 0; k < 3; ++k) ins(cs[base + k], ci[base + k]);
  }
  for (int k = 0; k < 3; ++k) { ss[tid * 3 + k] = bs[k]; si[tid * 3 + k] = bi[k]; }
  __syncthreads();
  if (tid == 0) {
    float fs[3] = {-3.4e38f, -3.4e38f, -3.4e38f}; int fi[3] = {0, 0, 0};
    for (int i = 0; i < 256 * 3; ++i) {
      float s = ss[i]; int id = si[i];
      if (s > fs[0] || (s == fs[0] && id < fi[0])) {
        fs[2] = fs[1]; fi[2] = fi[1]; fs[1] = fs[0]; fi[1] = fi[0]; fs[0] = s; fi[0] = id;
      } else if (s > fs[1] || (s == fs[1] && id < fi[1])) {
        fs[2] = fs[1]; fi[2] = fi[1]; fs[1] = s; fi[1] = id;
      } else if (s > fs[2] || (s == fs[2] && id < fi[2])) { fs[2] = s; fi[2] = id; }
    }
    for (int k = 0; k < 3; ++k) { ts[qi * 3 + k] = fs[k]; ti[qi * 3 + k] = fi[k]; }
  }
}

// ---------------- hcxr build ----------------
__global__ __launch_bounds__(256) void k_build_hcxr(const float* __restrict__ db,
                                                    const int* __restrict__ dblab,
                                                    const float* __restrict__ q,
                                                    const float* __restrict__ ts,
                                                    const int* __restrict__ ti,
                                                    float* __restrict__ hcxr,
                                                    int* __restrict__ labels) {
  int j = blockIdx.x;  // 0..95
  int b = j / Kc;
  int tid = threadIdx.x;
  int id = ti[j];
  float sc = ts[j];
  int lab = dblab[id];
  __shared__ float red[256];
  const float* dr = db + (size_t)id * Ec;
  float s = 0.f;
  for (int e = tid; e < Ec; e += 256) { float v = dr[e]; s += v * v; }
  red[tid] = s; __syncthreads();
  for (int st = 128; st > 0; st >>= 1) { if (tid < st) red[tid] += red[tid + st]; __syncthreads(); }
  float rinv = rsqrtf(red[0]);
  for (int e = tid; e < Ec; e += 256) {
    hcxr[((size_t)j * 3 + 0) * Ec + e] = (e == lab) ? sc : 0.f;
    hcxr[((size_t)j * 3 + 1) * Ec + e] = q[(size_t)b * Ec + e];
    hcxr[((size_t)j * 3 + 2) * Ec + e] = dr[e] * rinv;
  }
  if (tid == 0) labels[j] = lab;
}

// ---------------- mea attention (full-E dot, 3 tokens) ----------------
__global__ __launch_bounds__(256) void k_attn3(const float* __restrict__ Q,
                                               const float* __restrict__ Kb,
                                               const float* __restrict__ Vb,
                                               float* __restrict__ Ob) {
  int j = blockIdx.x, tid = threadIdx.x;
  __shared__ float sc[3][3];
  const float* q = Q + (size_t)j * 3 * Ec;
  const float* k = Kb + (size_t)j * 3 * Ec;
  const float* v = Vb + (size_t)j * 3 * Ec;
  if (tid < 9) {
    int qi = tid / 3, ki = tid % 3;
    float s = 0.f;
    for (int e = 0; e < Ec; ++e) s += q[qi * Ec + e] * k[ki * Ec + e];
    sc[qi][ki] = s * INV_SQRT_DK;
  }
  __syncthreads();
  if (tid < 3) {
    float m = fmaxf(sc[tid][0], fmaxf(sc[tid][1], sc[tid][2]));
    float e0 = expf(sc[tid][0] - m), e1 = expf(sc[tid][1] - m), e2 = expf(sc[tid][2] - m);
    float inv = 1.f / (e0 + e1 + e2);
    sc[tid][0] = e0 * inv; sc[tid][1] = e1 * inv; sc[tid][2] = e2 * inv;
  }
  __syncthreads();
  for (int t = tid; t < 3 * Ec; t += 256) {
    int qi = t / Ec, e = t % Ec;
    Ob[(size_t)j * 3 * Ec + t] =
        sc[qi][0] * v[e] + sc[qi][1] * v[Ec + e] + sc[qi][2] * v[2 * Ec + e];
  }
}

// ---------------- mea final / combine ----------------
__global__ void k_mea_final(const float* __restrict__ lg96, const int* __restrict__ labels,
                            float* __restrict__ mea) {
  int b = blockIdx.x, c = threadIdx.x;
  if (c >= NLc) return;
  float s = 0.f;
  for (int k = 0; k < Kc; ++k) s += lg96[(size_t)(b * Kc + k) * NLc + c];
  s /= (float)Kc;
  int cnt = 0;
  for (int k = 0; k < Kc; ++k) if (labels[b * Kc + k] == c) cnt++;
  float retr = (float)cnt / (float)Kc;
  mea[(size_t)b * NLc + c] = s * (1.f - RATIOc) + retr * RATIOc;
}

__global__ void k_combine(const float* __restrict__ ml, const float* __restrict__ me,
                          float* __restrict__ tgt, float* __restrict__ clsp) {
  int b = blockIdx.x, tid = threadIdx.x;
  __shared__ float lg[NLc];
  if (tid == 0) {
    float v[NLc]; float m = -3.4e38f;
    for (int c = 0; c < NLc; ++c) { v[c] = ml[b * NLc + c] + me[b * NLc + c]; m = fmaxf(m, v[c]); }
    float s = 0.f;
    for (int c = 0; c < NLc; ++c) { v[c] = expf(v[c] - m); s += v[c]; }
    int arg = 0; float best = -1.f;
    for (int c = 0; c < NLc; ++c) { v[c] /= s; if (v[c] > best) { best = v[c]; arg = c; } }
    for (int c = 0; c < NLc; ++c) lg[c] = v[c];
    clsp[b] = (float)(arg + 1);
  }
  __syncthreads();
  for (int e = tid; e < Ec; e += blockDim.x) tgt[(size_t)b * Ec + e] = (e < NLc) ? lg[e] : 0.f;
}

// ---------------- calibrator cross attention ----------------
__global__ __launch_bounds__(256) void k_ca_attn(const float* __restrict__ Qx,
                                                 const float* __restrict__ Kp,
                                                 const float* __restrict__ Vp,
                                                 float* __restrict__ ctx) {
  int blk = blockIdx.x;
  int b = blk >> 3, h = blk & 7;
  int tid = threadIdx.x;
  __shared__ float qv[80];
  __shared__ float p[Lc];
  __shared__ float red[256];
  if (tid < 80) qv[tid] = Qx[(size_t)b * Ec + h * 80 + tid];
  __syncthreads();
  for (int l = tid; l < Lc; l += 256) {
    const float* kr = Kp + ((size_t)b * Lc + l) * Ec + h * 80;
    float s = 0.f;
#pragma unroll 8
    for (int d = 0; d < 80; ++d) s += qv[d] * kr[d];
    p[l] = s * INV_SQRT_DK;
  }
  __syncthreads();
  float m = -3.4e38f;
  for (int l = tid; l < Lc; l += 256) m = fmaxf(m, p[l]);
  red[tid] = m; __syncthreads();
  for (int st = 128; st > 0; st >>= 1) { if (tid < st) red[tid] = fmaxf(red[tid], red[tid + st]); __syncthreads(); }
  m = red[0]; __syncthreads();
  float s = 0.f;
  for (int l = tid; l < Lc; l += 256) { float e = expf(p[l] - m); p[l] = e; s += e; }
  red[tid] = s; __syncthreads();
  for (int st = 128; st > 0; st >>= 1) { if (tid < st) red[tid] += red[tid + st]; __syncthreads(); }
  float inv = 1.f / red[0]; __syncthreads();
  if (tid < 80) {
    float acc = 0.f;
    for (int l = 0; l < Lc; ++l) acc += p[l] * Vp[((size_t)b * Lc + l) * Ec + h * 80 + tid];
    ctx[(size_t)b * Ec + h * 80 + tid] = acc * inv;
  }
}

// ---------------- final output ----------------
__global__ void k_final(const float* __restrict__ xcal, const float* __restrict__ lw,
                        const float* __restrict__ lb, const float* __restrict__ clsp,
                        float* __restrict__ out) {
  int b = threadIdx.x;
  if (b >= Bc) return;
  float s = lb[0];
  for (int e = 0; e < Ec; ++e) s += xcal[(size_t)b * Ec + e] * lw[e];
  if (s != s) s = 0.f;
  else if (s > 3.402823466e38f) s = 3.402823466e38f;
  else if (s < -3.402823466e38f) s = -3.402823466e38f;
  out[b] = clsp[b] + s;
}

// ================= host side =================
static inline int cdiv(long a, long b) { return (int)((a + b - 1) / b); }

extern "C" void kernel_launch(void* const* d_in, const int* in_sizes, int n_in,
                              void* d_out, int out_size, void* d_ws, size_t ws_size,
                              hipStream_t stream) {
  if (n_in < 183) return;
  auto F = [&](int i) { return (const float*)d_in[i]; };
  const float* emb = F(0);
  const float* db = F(1);
  const int* dblab = (const int*)d_in[2];
  // index map (insertion order, recursive)
  const int MR_V = 3, MR_W = 21;                // conv blocks (18 each)
  const int LNG = 39, LNB = 40, MDW = 41, MDB = 42, MOW = 43, MOB = 44;
  const int MEA = 45;                            // 3 layers x 8
  const int MEA_DW = 69, MEA_DB = 70, MEA_OW = 71, MEA_OB = 72;
  const int CAL = 73;                            // 6 layers x 18
  const int LW = 181, LB = 182;

  size_t off = 0;
  auto alloc = [&](size_t bytes) -> void* {
    void* p = (char*)d_ws + off;
    off = (off + bytes + 255) & ~(size_t)255;
    return p;
  };
  __bf16* xT = (__bf16*)alloc((size_t)Bc * Ec * Lc * 2);
  __bf16* t1 = (__bf16*)alloc((size_t)Bc * MIDc * Lc * 2);
  __bf16* t2 = (__bf16*)alloc((size_t)Bc * MIDc * Lc * 2);
  float* valsF = (float*)alloc((size_t)Bc * Ec * Lc * 4);
  float* wtsF = (float*)alloc((size_t)Bc * Ec * Lc * 4);
  __bf16* pc1 = (__bf16*)alloc((size_t)MIDc * Ec * 3 * 2);
  __bf16* pr1 = (__bf16*)alloc((size_t)MIDc * Ec * 2);
  __bf16* pc2 = (__bf16*)alloc((size_t)MIDc * MIDc * 5 * 2);
  __bf16* pr2 = (__bf16*)alloc((size_t)MIDc * MIDc * 2);
  __bf16* pc3 = (__bf16*)alloc((size_t)Ec * MIDc * 7 * 2);
  __bf16* pr3 = (__bf16*)alloc((size_t)Ec * MIDc * 2);
  float* xs = (float*)alloc((size_t)Bc * Ec * 4);
  float* xm = (float*)alloc((size_t)Bc * Ec * 4);
  float* hbuf = (float*)alloc((size_t)Bc * 2 * Ec * 4);
  float* h2 = (float*)alloc((size_t)Bc * 2 * Ec * 4);
  float* mrl = (float*)alloc((size_t)Bc * NLc * 4);
  float* qbuf = (float*)alloc((size_t)Bc * Ec * 4);
  float* qn = (float*)alloc((size_t)Bc * 4);
  float* cs = (float*)alloc((size_t)TKB * Bc * 3 * 4);
  int* ci = (int*)alloc((size_t)TKB * Bc * 3 * 4);
  float* ts = (float*)alloc((size_t)Bc * 3 * 4);
  int* ti = (int*)alloc((size_t)Bc * 3 * 4);
  int* labs = (int*)alloc((size_t)Bc * 3 * 4);
  float* hA = (float*)alloc((size_t)96 * 3 * Ec * 4);
  float* hB = (float*)alloc((size_t)96 * 3 * Ec * 4);
  float* qp = (float*)alloc((size_t)288 * Ec * 4);
  float* kp = (float*)alloc((size_t)288 * Ec * 4);
  float* vp = (float*)alloc((size_t)288 * Ec * 4);
  float* ap = (float*)alloc((size_t)288 * Ec * 4);
  float* feat2 = (float*)alloc((size_t)96 * Ec * 4);
  float* lg96 = (float*)alloc((size_t)96 * NLc * 4);
  float* meaL = (float*)alloc((size_t)Bc * NLc * 4);
  float* xcal = (float*)alloc((size_t)Bc * Ec * 4);
  float* clsp = (float*)alloc((size_t)Bc * 4);
  __bf16* embB = (__bf16*)alloc((size_t)Bc * Lc * Ec * 2);
  __bf16* cawB = (__bf16*)alloc((size_t)3 * Ec * Ec * 2);
  float* kproj = (float*)alloc((size_t)Bc * Lc * Ec * 4);
  float* vproj = (float*)alloc((size_t)Bc * Lc * Ec * 4);
  float* sav = (float*)alloc((size_t)Bc * Ec * 4);
  float* sao = (float*)alloc((size_t)Bc * Ec * 4);
  float* qx = (float*)alloc((size_t)Bc * Ec * 4);
  float* ctx = (float*)alloc((size_t)Bc * Ec * 4);
  float* cao = (float*)alloc((size_t)Bc * Ec * 4);
  float* ff1 = (float*)alloc((size_t)Bc * FFc * 4);
  float* ff2 = (float*)alloc((size_t)Bc * Ec * 4);
  (void)ws_size; (void)in_sizes; (void)out_size;

  auto gemm = [&](const float* A, const float* W, const float* bias, float* C,
                  int M, int N, int Kd, int lda, int act) {
    k_gemm_small<<<cdiv((long)M * N, 256), 256, 0, stream>>>(A, W, bias, C, M, N, Kd, lda, act);
  };

  // ---- MRLAT ----
  {
    long n = (long)Bc * Ec * Lc;
    k_emb_to_xT<<<cdiv(n, 256), 256, 0, stream>>>(emb, xT);
  }
  for (int blk = 0; blk < 2; ++blk) {
    int base = (blk == 0) ? MR_V : MR_W;
    k_pack_w<<<cdiv((long)MIDc * Ec * 3, 256), 256, 0, stream>>>(F(base + 0), pc1, MIDc, Ec, 3);
    k_pack_w<<<cdiv((long)MIDc * Ec, 256), 256, 0, stream>>>(F(base + 6), pr1, MIDc, Ec, 1);
    k_pack_w<<<cdiv((long)MIDc * MIDc * 5, 256), 256, 0, stream>>>(F(base + 2), pc2, MIDc, MIDc, 5);
    k_pack_w<<<cdiv((long)MIDc * MIDc, 256), 256, 0, stream>>>(F(base + 8), pr2, MIDc, MIDc, 1);
    k_pack_w<<<cdiv((long)Ec * MIDc * 7, 256), 256, 0, stream>>>(F(base + 4), pc3, Ec, MIDc, 7);
    k_pack_w<<<cdiv((long)Ec * MIDc, 256), 256, 0, stream>>>(F(base + 10), pr3, Ec, MIDc, 1);
    dim3 g1(Lc / 128, MIDc / 64, Bc);
    k_conv_wmma<<<g1, 256, 0, stream>>>(xT, pc1, pr1, F(base + 1), F(base + 7),
                                        F(base + 12), F(base + 13), t1, nullptr,
                                        Ec, MIDc, Lc, 3, 1);
    dim3 g2(Lc / 128, MIDc / 64, Bc);
    k_conv_wmma<<<g2, 256, 0, stream>>>(t1, pc2, pr2, F(base + 3), F(base + 9),
                                        F(base + 14), F(base + 15), t2, nullptr,
                                        MIDc, MIDc, Lc, 5, 2);
    dim3 g3(Lc / 128, Ec / 64, Bc);
    k_conv_wmma<<<g3, 256, 0, stream>>>(t2, pc3, pr3, F(base + 5), F(base + 11),
                                        F(base + 16), F(base + 17), nullptr,
                                        blk == 0 ? valsF : wtsF, MIDc, Ec, Lc, 7, 3);
  }
  k_pool<<<cdiv((long)Bc * Ec, 8), 256, 0, stream>>>(valsF, wtsF, xs, xm);
  k_ln_to_h<<<Bc, 256, 0, stream>>>(xs, xm, F(LNG), F(LNB), hbuf);
  gemm(hbuf, F(MDW), F(MDB), h2, Bc, 2 * Ec, 2 * Ec, 2 * Ec, 1);
  gemm(h2, F(MOW), F(MOB), mrl, Bc, NLc, 2 * Ec, 2 * Ec, 0);

  // ---- MEA retrieval ----
  k_qmean<<<Bc, 256, 0, stream>>>(emb, qbuf);
  k_rownorm<<<Bc, 256, 0, stream>>>(qbuf, qn, Ec);
  k_sims_topk<<<TKB, 256, 0, stream>>>(db, qbuf, qn, cs, ci);
  k_topk_merge<<<Bc, 256, 0, stream>>>(cs, ci, ts, ti, TKB);
  k_build_hcxr<<<96, 256, 0, stream>>>(db, dblab, qbuf, ts, ti, hA, labs);

  float* cur = hA; float* nxt = hB;
  for (int l = 0; l < MLc; ++l) {
    int b = MEA + 8 * l;
    gemm(cur, F(b + 0), F(b + 1), qp, 288, Ec, Ec, Ec, 0);
    gemm(cur, F(b + 2), F(b + 3), kp, 288, Ec, Ec, Ec, 0);
    gemm(cur, F(b + 4), F(b + 5), vp, 288, Ec, Ec, Ec, 0);
    k_attn3<<<96, 256, 0, stream>>>(qp, kp, vp, ap);
    gemm(ap, F(b + 6), F(b + 7), nxt, 288, Ec, Ec, Ec, 0);
    float* tmp = cur; cur = nxt; nxt = tmp;
  }
  // feat = hcxr[:,0,:]  (row stride 3*E)
  gemm(cur, F(MEA_DW), F(MEA_DB), feat2, 96, Ec, Ec, 3 * Ec, 1);
  gemm(feat2, F(MEA_OW), F(MEA_OB), lg96, 96, NLc, Ec, Ec, 0);
  k_mea_final<<<Bc, 32, 0, stream>>>(lg96, labs, meaL);
  k_combine<<<Bc, 64, 0, stream>>>(mrl, meaL, xcal, clsp);

  // ---- calibrator ----
  k_f32_to_bf16<<<cdiv((long)Bc * Lc * Ec, 256), 256, 0, stream>>>(emb, embB, (long)Bc * Lc * Ec);
  for (int l = 0; l < CLc; ++l) {
    int b = CAL + 18 * l;
    const float* saw = F(b + 0);  const float* sab = F(b + 1);
    const float* saow = F(b + 2); const float* saob = F(b + 3);
    const float* caw = F(b + 4);  const float* cab = F(b + 5);
    const float* caow = F(b + 6); const float* caob = F(b + 7);
    // self-attn on single token: softmax(1x1)==1 -> out = (x@wv^T+bv)@saow^T+saob
    gemm(xcal, saw + (size_t)2 * Ec * Ec, sab + 2 * Ec, sav, Bc, Ec, Ec, Ec, 0);
    gemm(sav, saow, saob, sao, Bc, Ec, Ec, Ec, 0);
    k_ln_res<<<Bc, 256, 0, stream>>>(xcal, sao, F(b + 12), F(b + 13));
    // cross-attn: Q from x; K,V from emb via WMMA GEMM (M=B*L)
    k_f32_to_bf16<<<cdiv((long)3 * Ec * Ec, 256), 256, 0, stream>>>(caw, cawB, (long)3 * Ec * Ec);
    gemm(xcal, caw, cab, qx, Bc, Ec, Ec, Ec, 0);
    dim3 gg(Ec / 64, (Bc * Lc) / 128);
    k_gemm_wmma<<<gg, 256, 0, stream>>>(embB, cawB + (size_t)Ec * Ec, cab + Ec,
                                        kproj, Bc * Lc, Ec, Ec);
    k_gemm_wmma<<<gg, 256, 0, stream>>>(embB, cawB + (size_t)2 * Ec * Ec, cab + 2 * Ec,
                                        vproj, Bc * Lc, Ec, Ec);
    k_ca_attn<<<Bc * Hc, 256, 0, stream>>>(qx, kproj, vproj, ctx);
    gemm(ctx, caow, caob, cao, Bc, Ec, Ec, Ec, 0);
    k_ln_res<<<Bc, 256, 0, stream>>>(xcal, cao, F(b + 14), F(b + 15));
    // FF
    gemm(xcal, F(b + 8), F(b + 9), ff1, Bc, FFc, Ec, Ec, 2);
    gemm(ff1, F(b + 10), F(b + 11), ff2, Bc, Ec, FFc, FFc, 0);
    k_ln_res<<<Bc, 256, 0, stream>>>(xcal, ff2, F(b + 16), F(b + 17));
  }
  k_final<<<1, 32, 0, stream>>>(xcal, F(LW), F(LB), clsp, (float*)d_out);
}
